// GCNNet_27187142983949
// MI455X (gfx1250) — matrix operand order repaired and checked by reading, not compiled, not run on hardware
//
#include <hip/hip_runtime.h>
#include <hip/hip_bf16.h>
#include <stdint.h>

typedef __attribute__((ext_vector_type(16))) _Float16 v16h;
typedef __attribute__((ext_vector_type(8)))  _Float16 v8h;
typedef __attribute__((ext_vector_type(8)))  float    v8f;
typedef __attribute__((ext_vector_type(4)))  float    v4f;

#define N_NODES 100000
#define N_EDGES 600000
#define F_IN    128
#define HID     256
#define NCLS    64

__device__ __forceinline__ void atomic_add_f32(float* p, float v) {
    unsafeAtomicAdd(p, v);   // hardware global_atomic_add_f32 path
}

// ---------------------------------------------------------------------------
// Convert + transpose weights to f16 once per launch:
//   W1 [128x256] -> W1t [256x128] f16 ; W2 [256x64] -> W2t [64x256] f16
// ---------------------------------------------------------------------------
__global__ void prep_weights(const float* __restrict__ W1, const float* __restrict__ W2,
                             _Float16* __restrict__ W1t, _Float16* __restrict__ W2t) {
    int i = blockIdx.x * blockDim.x + threadIdx.x;
    if (i < F_IN * HID) {
        int k = i / HID, n = i % HID;
        W1t[n * F_IN + k] = (_Float16)W1[i];
    }
    int j = i - F_IN * HID;
    if (j >= 0 && j < HID * NCLS) {
        int k = j / NCLS, n = j % NCLS;
        W2t[n * HID + k] = (_Float16)W2[j];
    }
}

// deg = 1 (self loop), agg buffers = 0  (must re-init every call: graph replay)
__global__ void init_buffers(float* __restrict__ deg, float* __restrict__ agg1,
                             float* __restrict__ agg2) {
    const int64_t n1 = N_NODES;
    const int64_t n2 = n1 + (int64_t)N_NODES * HID;
    const int64_t total = n2 + (int64_t)N_NODES * NCLS;
    for (int64_t i = (int64_t)blockIdx.x * blockDim.x + threadIdx.x; i < total;
         i += (int64_t)gridDim.x * blockDim.x) {
        if (i < n1)      deg[i] = 1.0f;
        else if (i < n2) agg1[i - n1] = 0.0f;
        else             agg2[i - n2] = 0.0f;
    }
}

__global__ void deg_accum(const int* __restrict__ dst, float* __restrict__ deg) {
    int e = blockIdx.x * blockDim.x + threadIdx.x;
    if (e < N_EDGES) atomic_add_f32(&deg[dst[e]], 1.0f);
}

__global__ void compute_dinv(float* __restrict__ deg_dinv) {
    int n = blockIdx.x * blockDim.x + threadIdx.x;
    if (n < N_NODES) deg_dinv[n] = rsqrtf(deg_dinv[n]);
}

// ---------------------------------------------------------------------------
// GEMM1: h1 = x[N x 128] @ W1[128 x 256]  -> f16 [N x 256]
// block = 256 thr (8 waves). Block owns a 16-row M tile; wave w owns cols
// [32w, 32w+32) as two 16x16 WMMA tiles. K loop: 4 steps of 32.
// ---------------------------------------------------------------------------
__global__ void __launch_bounds__(256) gemm1_wmma(const float* __restrict__ x,
                                                  const _Float16* __restrict__ W1t,
                                                  _Float16* __restrict__ h1h) {
    const int wave = threadIdx.x >> 5;
    const int lane = threadIdx.x & 31;
    const int mrow = lane & 15;
    const int hi   = lane >> 4;              // half-wave select
    const int row  = blockIdx.x * 16 + mrow; // N divisible by 16
    const int colbase = wave * 32;

    v8f c0 = {}; v8f c1 = {};
    const float*    xrow = x   + (size_t)row * F_IN;
    const _Float16* bp0  = W1t + (size_t)(colbase + mrow)      * F_IN;
    const _Float16* bp1  = W1t + (size_t)(colbase + 16 + mrow) * F_IN;

#pragma unroll
    for (int ks = 0; ks < 4; ++ks) {
        const int kofs  = ks * 32;
        const int kbase = hi * 8;            // A: K = {kbase..kbase+7, 16+kbase..}
        v16h a;
        {
            v4f t0 = *(const v4f*)(xrow + kofs + kbase);
            v4f t1 = *(const v4f*)(xrow + kofs + kbase + 4);
            v4f t2 = *(const v4f*)(xrow + kofs + 16 + kbase);
            v4f t3 = *(const v4f*)(xrow + kofs + 16 + kbase + 4);
#pragma unroll
            for (int j = 0; j < 4; ++j) {
                a[j]      = (_Float16)t0[j];
                a[j + 4]  = (_Float16)t1[j];
                a[j + 8]  = (_Float16)t2[j];
                a[j + 12] = (_Float16)t3[j];
            }
        }
        const int kb = hi * 8;               // B: K = {kb..kb+7, 16+kb..16+kb+7}
        v16h b0, b1;
        {
            v8h p0 = *(const v8h*)(bp0 + kofs + kb);
            v8h p1 = *(const v8h*)(bp0 + kofs + 16 + kb);
            v8h q0 = *(const v8h*)(bp1 + kofs + kb);
            v8h q1 = *(const v8h*)(bp1 + kofs + 16 + kb);
#pragma unroll
            for (int j = 0; j < 8; ++j) {
                b0[j] = p0[j]; b0[j + 8] = p1[j];
                b1[j] = q0[j]; b1[j + 8] = q1[j];
            }
        }
        c0 = __builtin_amdgcn_wmma_f32_16x16x32_f16(false, a, false, b0, (short)0, c0, false, false);
        c1 = __builtin_amdgcn_wmma_f32_16x16x32_f16(false, a, false, b1, (short)0, c1, false, false);
    }

    const int rbase = blockIdx.x * 16 + hi * 8; // C/D layout: hi selects M 0-7 / 8-15
#pragma unroll
    for (int v = 0; v < 8; ++v) {
        h1h[(size_t)(rbase + v) * HID + colbase + mrow]      = (_Float16)c0[v];
        h1h[(size_t)(rbase + v) * HID + colbase + 16 + mrow] = (_Float16)c1[v];
    }
}

// One wave per edge: lane handles 8 of 256 features. Gather f16 (L2-resident),
// scale, hardware f32 scatter-atomics.
__global__ void __launch_bounds__(256) aggregate1(const int* __restrict__ src,
                                                  const int* __restrict__ dst,
                                                  const float* __restrict__ dinv,
                                                  const _Float16* __restrict__ h1h,
                                                  float* __restrict__ agg1) {
    const int e = (blockIdx.x * blockDim.x + threadIdx.x) >> 5;
    if (e >= N_EDGES) return;
    const int lane = threadIdx.x & 31;
    const int s = src[e], d = dst[e];
    const float norm = dinv[s] * dinv[d];
    const v8h hv = *(const v8h*)(h1h + (size_t)s * HID + lane * 8);
    float* out = agg1 + (size_t)d * HID + lane * 8;
#pragma unroll
    for (int j = 0; j < 8; ++j) atomic_add_f32(out + j, (float)hv[j] * norm);
}

// hrelu = relu(agg1 + h1*dinv^2 + b1) -> f16
__global__ void finalize1(const float* __restrict__ agg1, const _Float16* __restrict__ h1h,
                          const float* __restrict__ dinv, const float* __restrict__ b1,
                          _Float16* __restrict__ hrelu) {
    int64_t i = (int64_t)blockIdx.x * blockDim.x + threadIdx.x;
    if (i >= (int64_t)N_NODES * HID) return;
    const int n = (int)(i >> 8);
    const int f = (int)(i & 255);
    float sw = dinv[n]; sw *= sw;
    float v = agg1[i] + (float)h1h[i] * sw + b1[f];
    hrelu[i] = (_Float16)fmaxf(v, 0.0f);
}

// ---------------------------------------------------------------------------
// GEMM2: h2 = hrelu[N x 256] @ W2[256 x 64] -> f16 [N x 64]
// block = 128 thr (4 waves); wave w owns 16-col tile w; K loop: 8 steps of 32.
// ---------------------------------------------------------------------------
__global__ void __launch_bounds__(128) gemm2_wmma(const _Float16* __restrict__ hrelu,
                                                  const _Float16* __restrict__ W2t,
                                                  _Float16* __restrict__ h2h) {
    const int wave = threadIdx.x >> 5;
    const int lane = threadIdx.x & 31;
    const int mrow = lane & 15;
    const int hi   = lane >> 4;
    const int row  = blockIdx.x * 16 + mrow;
    const int colbase = wave * 16;

    v8f c = {};
    const _Float16* arow = hrelu + (size_t)row * HID;
    const _Float16* bp   = W2t  + (size_t)(colbase + mrow) * HID;

#pragma unroll
    for (int ks = 0; ks < 8; ++ks) {
        const int kofs  = ks * 32;
        const int kbase = hi * 8;
        v16h a, b;
        {
            v8h a0 = *(const v8h*)(arow + kofs + kbase);
            v8h a1 = *(const v8h*)(arow + kofs + 16 + kbase);
            v8h p0 = *(const v8h*)(bp + kofs + kbase);
            v8h p1 = *(const v8h*)(bp + kofs + 16 + kbase);
#pragma unroll
            for (int j = 0; j < 8; ++j) {
                a[j] = a0[j]; a[j + 8] = a1[j];
                b[j] = p0[j]; b[j + 8] = p1[j];
            }
        }
        c = __builtin_amdgcn_wmma_f32_16x16x32_f16(false, a, false, b, (short)0, c, false, false);
    }

    const int rbase = blockIdx.x * 16 + hi * 8;
#pragma unroll
    for (int v = 0; v < 8; ++v)
        h2h[(size_t)(rbase + v) * NCLS + colbase + mrow] = (_Float16)c[v];
}

// One wave per edge: lane handles 2 of 64 features.
__global__ void __launch_bounds__(256) aggregate2(const int* __restrict__ src,
                                                  const int* __restrict__ dst,
                                                  const float* __restrict__ dinv,
                                                  const _Float16* __restrict__ h2h,
                                                  float* __restrict__ agg2) {
    const int e = (blockIdx.x * blockDim.x + threadIdx.x) >> 5;
    if (e >= N_EDGES) return;
    const int lane = threadIdx.x & 31;
    const int s = src[e], d = dst[e];
    const float norm = dinv[s] * dinv[d];
    const _Float16* hp = h2h + (size_t)s * NCLS + lane * 2;
    float* out = agg2 + (size_t)d * NCLS + lane * 2;
    atomic_add_f32(out,     (float)hp[0] * norm);
    atomic_add_f32(out + 1, (float)hp[1] * norm);
}

// logits = agg2 + h2*dinv^2 + b2 ; out = log_softmax(logits) ; wave per node
__global__ void __launch_bounds__(256) finalize2(const float* __restrict__ agg2,
                                                 const _Float16* __restrict__ h2h,
                                                 const float* __restrict__ dinv,
                                                 const float* __restrict__ b2,
                                                 float* __restrict__ out) {
    const int n = (blockIdx.x * blockDim.x + threadIdx.x) >> 5;
    if (n >= N_NODES) return;
    const int lane = threadIdx.x & 31;
    float sw = dinv[n]; sw *= sw;
    const size_t base = (size_t)n * NCLS;
    float v0 = agg2[base + lane]      + (float)h2h[base + lane]      * sw + b2[lane];
    float v1 = agg2[base + 32 + lane] + (float)h2h[base + 32 + lane] * sw + b2[32 + lane];
    float m = fmaxf(v0, v1);
#pragma unroll
    for (int off = 16; off > 0; off >>= 1) m = fmaxf(m, __shfl_xor(m, off, 32));
    float s = expf(v0 - m) + expf(v1 - m);
#pragma unroll
    for (int off = 16; off > 0; off >>= 1) s += __shfl_xor(s, off, 32);
    const float lse = m + logf(s);
    out[base + lane]      = v0 - lse;
    out[base + 32 + lane] = v1 - lse;
}

// ---------------------------------------------------------------------------
extern "C" void kernel_launch(void* const* d_in, const int* in_sizes, int n_in,
                              void* d_out, int out_size, void* d_ws, size_t ws_size,
                              hipStream_t stream) {
    const float* x  = (const float*)d_in[0];
    const float* W1 = (const float*)d_in[1];
    const float* b1 = (const float*)d_in[2];
    const float* W2 = (const float*)d_in[3];
    const float* b2 = (const float*)d_in[4];
    const int*   ei = (const int*)d_in[5];   // [2, E]
    const int* srcp = ei;
    const int* dstp = ei + N_EDGES;

    // workspace carve (256B aligned)
    char* ws = (char*)d_ws;
    auto take = [&](size_t bytes) -> char* {
        char* p = ws;
        ws += (bytes + 255) & ~(size_t)255;
        return p;
    };
    float*     dinv  = (float*)take((size_t)N_NODES * 4);
    _Float16*  h1h   = (_Float16*)take((size_t)N_NODES * HID * 2);
    float*     agg1  = (float*)take((size_t)N_NODES * HID * 4);
    _Float16*  hrelu = (_Float16*)take((size_t)N_NODES * HID * 2);
    _Float16*  h2h   = (_Float16*)take((size_t)N_NODES * NCLS * 2);
    float*     agg2  = (float*)take((size_t)N_NODES * NCLS * 4);
    _Float16*  W1t   = (_Float16*)take((size_t)F_IN * HID * 2);
    _Float16*  W2t   = (_Float16*)take((size_t)HID * NCLS * 2);

    prep_weights<<<(F_IN * HID + HID * NCLS + 255) / 256, 256, 0, stream>>>(W1, W2, W1t, W2t);
    init_buffers<<<4096, 256, 0, stream>>>(dinv, agg1, agg2);
    deg_accum<<<(N_EDGES + 255) / 256, 256, 0, stream>>>(dstp, dinv);
    compute_dinv<<<(N_NODES + 255) / 256, 256, 0, stream>>>(dinv);

    gemm1_wmma<<<N_NODES / 16, 256, 0, stream>>>(x, W1t, h1h);
    aggregate1<<<(N_EDGES * 32 + 255) / 256, 256, 0, stream>>>(srcp, dstp, dinv, h1h, agg1);
    finalize1<<<(N_NODES * HID + 255) / 256, 256, 0, stream>>>(agg1, h1h, dinv, b1, hrelu);

    gemm2_wmma<<<N_NODES / 16, 128, 0, stream>>>(hrelu, W2t, h2h);
    aggregate2<<<(N_EDGES * 32 + 255) / 256, 256, 0, stream>>>(srcp, dstp, dinv, h2h, agg2);
    finalize2<<<(N_NODES * 32 + 255) / 256, 256, 0, stream>>>(agg2, h2h, dinv, b2, (float*)d_out);
}
